// MultiHeadedAttentionExport_6717328851573
// MI455X (gfx1250) — matrix operand order repaired and checked by reading, not compiled, not run on hardware
//
#include <hip/hip_runtime.h>
#include <hip/hip_bf16.h>

// Problem constants (match reference)
#define Bc 4
#define Tc 2048
#define Dc 1024
#define Hc 16
#define DKc 64

typedef __bf16 bf16_t;
typedef __attribute__((ext_vector_type(16))) __bf16 v16bf;
typedef __attribute__((ext_vector_type(8)))  __bf16 bf16x8;
typedef __attribute__((ext_vector_type(4)))  __bf16 bf16x4;
typedef __attribute__((ext_vector_type(8)))  float  v8f;

// ---------------------------------------------------------------------------
// WMMA helper: D = A(16x32 bf16) x B(32x16 bf16) + C (f32)
// ---------------------------------------------------------------------------
__device__ __forceinline__ v8f wmma_bf16(v16bf a, v16bf b, v8f c) {
    return __builtin_amdgcn_wmma_f32_16x16x32_bf16(
        /*neg_a=*/false, a, /*neg_b=*/false, b,
        /*c_mod=*/(short)0, c, /*reuse_a=*/false, /*reuse_b=*/false);
}

// A-fragment (16x32, 16-bit): lane holds row (l&15); halves 0..7 = K kb..kb+7,
// halves 8..15 = K kb+16..kb+23, kb = 8*(l>=16). rowp -> start of 32-elem K chunk.
__device__ __forceinline__ v16bf load_afrag(const bf16_t* rowp, int hi) {
    const int kb = hi * 8;
    bf16x8 lo = *(const bf16x8*)(rowp + kb);
    bf16x8 hp = *(const bf16x8*)(rowp + kb + 16);
    v16bf r;
#pragma unroll
    for (int i = 0; i < 8; ++i) { r[i] = lo[i]; r[i + 8] = hp[i]; }
    return r;
}

// B-fragment (32x16, 16-bit): lane holds column (l&15); halves 0..7 = K kb..kb+7,
// halves 8..15 = K kb+16..kb+23, kb = 8*(l>=16). colp -> start of the 32-elem K chunk.
__device__ __forceinline__ v16bf load_bfrag(const bf16_t* colp, int hi) {
    const int k0 = hi * 8;
    bf16x8 lo = *(const bf16x8*)(colp + k0);
    bf16x8 hp = *(const bf16x8*)(colp + k0 + 16);
    v16bf r;
#pragma unroll
    for (int i = 0; i < 8; ++i) { r[i] = lo[i]; r[i + 8] = hp[i]; }
    return r;
}

// ---------------------------------------------------------------------------
// fp32 -> bf16 conversion (vectorized x4)
// ---------------------------------------------------------------------------
__global__ __launch_bounds__(256) void cvt_bf16_kernel(
    const float* __restrict__ s, bf16_t* __restrict__ d, int n4) {
    int i = blockIdx.x * blockDim.x + threadIdx.x;
    if (i < n4) {
        float4 f = ((const float4*)s)[i];
        bf16x4 o;
        o[0] = (bf16_t)f.x; o[1] = (bf16_t)f.y;
        o[2] = (bf16_t)f.z; o[3] = (bf16_t)f.w;
        *(bf16x4*)(d + 4 * (size_t)i) = o;
    }
}

// ---------------------------------------------------------------------------
// Projection GEMM: Y[m,e] = sum_d X[m,d] * W[e,d] + bias[e]
//   X: [M=B*T, Dc] bf16   W: [Dc, Dc] bf16 (torch [out,in])   bias: f32
// One wave computes a 32x64 tile: 2 A-frags x 4 B-frags -> 8 WMMAs per K-step
// (B fragments reused across both M-tiles).
// MODE 0: store bf16 -> [B,H,T,DK]        (Q, K)
// MODE 1: store bf16 -> [B,H,DK,T]        (V transposed)
// MODE 2: store f32  -> [B*T, Dc]         (final output)
// ---------------------------------------------------------------------------
template <int MODE>
__global__ __launch_bounds__(256) void proj_gemm_kernel(
    const bf16_t* __restrict__ X, const bf16_t* __restrict__ W,
    const float* __restrict__ bias, void* __restrict__ outp) {
    const int lane = threadIdx.x & 31;
    const int ln   = lane & 15;
    const int hi   = lane >> 4;
    const int wid  = blockIdx.x * (blockDim.x >> 5) + (threadIdx.x >> 5);
    const int TN   = Dc / 64;              // 16 column blocks of 64
    const int tn   = wid % TN;
    const int tm   = wid / TN;
    const int m0   = tm * 32;
    const int n0   = tn * 64;

    v8f acc[2][4] = {{v8f{}, v8f{}, v8f{}, v8f{}}, {v8f{}, v8f{}, v8f{}, v8f{}}};

    const bf16_t* xr0 = X + (size_t)(m0 + ln) * Dc;
    const bf16_t* xr1 = X + (size_t)(m0 + 16 + ln) * Dc;
    for (int kk = 0; kk < Dc; kk += 32) {
        if (kk + 32 < Dc) {
            __builtin_prefetch(xr0 + kk + 32, 0, 0);
            __builtin_prefetch(xr1 + kk + 32, 0, 0);
        }
        v16bf a0 = load_afrag(xr0 + kk, hi);
        v16bf a1 = load_afrag(xr1 + kk, hi);
#pragma unroll
        for (int j = 0; j < 4; ++j) {
            const int e = n0 + j * 16 + ln;
            v16bf b = load_bfrag(W + (size_t)e * Dc + kk, hi);
            acc[0][j] = wmma_bf16(a0, b, acc[0][j]);
            acc[1][j] = wmma_bf16(a1, b, acc[1][j]);
        }
    }

    // bias (per-lane column is fixed within each 16-wide tile)
#pragma unroll
    for (int j = 0; j < 4; ++j) {
        const float bv = bias[n0 + j * 16 + ln];
#pragma unroll
        for (int r = 0; r < 8; ++r) { acc[0][j][r] += bv; acc[1][j][r] += bv; }
    }

    // C/D layout: vgpr r, lanes 0-15 -> M=r, lanes 16-31 -> M=8+r; N=l&15.
#pragma unroll
    for (int i = 0; i < 2; ++i) {
        const int mi = m0 + 16 * i;
        if (MODE == 0) {
            bf16_t* out = (bf16_t*)outp;
#pragma unroll
            for (int j = 0; j < 4; ++j) {
                const int e  = n0 + j * 16 + ln;
                const int h  = e >> 6;
                const int dk = e & 63;
#pragma unroll
                for (int r = 0; r < 8; ++r) {
                    const int m = mi + r + 8 * hi;
                    const int b = m >> 11;         // / T
                    const int t = m & (Tc - 1);
                    out[(((size_t)(b * Hc + h)) * Tc + t) * DKc + dk] =
                        (bf16_t)acc[i][j][r];
                }
            }
        } else if (MODE == 1) {
            bf16_t* out = (bf16_t*)outp;
            const int mb = mi + 8 * hi;            // 8 consecutive t per lane
            const int b  = mb >> 11;
            const int tb = mb & (Tc - 1);
#pragma unroll
            for (int j = 0; j < 4; ++j) {
                const int e  = n0 + j * 16 + ln;
                const int h  = e >> 6;
                const int dk = e & 63;
                bf16x8 pk;
#pragma unroll
                for (int r = 0; r < 8; ++r) pk[r] = (bf16_t)acc[i][j][r];
                *(bf16x8*)(out + (((size_t)(b * Hc + h)) * DKc + dk) * Tc + tb) = pk;
            }
        } else {
            float* out = (float*)outp;
#pragma unroll
            for (int j = 0; j < 4; ++j) {
                const int e = n0 + j * 16 + ln;
#pragma unroll
                for (int r = 0; r < 8; ++r) {
                    const int m = mi + r + 8 * hi;
                    out[(size_t)m * Dc + e] = acc[i][j][r];
                }
            }
        }
    }
}

// ---------------------------------------------------------------------------
// Flash attention: one wave handles 32 query rows (2 M-tiles) of one (b,h),
// sharing every K/V fragment across both tiles (2x arithmetic intensity).
//   Q,K: [B*H, T, DK] bf16   VT: [B*H, DK, T] bf16   mask: [B,1,T,T] f32
//   CTX out: [B, T, D] bf16
// Per 32-key chunk: 8 score WMMAs, online softmax (max via xor-shuffles,
// row-SUM via an extra WMMA against an all-ones B matrix -> lands in
// accumulator layout, no cross-lane ops), P staged via LDS, 8 P*V WMMAs
// + 2 row-sum WMMAs.
// ---------------------------------------------------------------------------
#define PROW 40  // padded LDS row (halves): kills 16x bank conflicts
__global__ __launch_bounds__(128) void flash_attn_kernel(
    const bf16_t* __restrict__ Q, const bf16_t* __restrict__ K,
    const bf16_t* __restrict__ VT, const float* __restrict__ mask,
    bf16_t* __restrict__ CTX) {
    __shared__ bf16_t pbuf[4][2][16 * PROW];

    const int lane  = threadIdx.x & 31;
    const int ln    = lane & 15;
    const int hi    = lane >> 4;
    const int wslot = threadIdx.x >> 5;
    const int wid   = blockIdx.x * 4 + wslot;
    const int bh    = wid >> 6;            // / (T/32)
    const int qt    = wid & 63;
    const int b     = bh >> 4;             // / H
    const int h     = bh & (Hc - 1);
    const int tq0   = qt * 32;

    const bf16_t* Qb = Q  + (size_t)bh * Tc * DKc;
    const bf16_t* Kb = K  + (size_t)bh * Tc * DKc;
    const bf16_t* Vb = VT + (size_t)bh * DKc * Tc;
    const float*  Mb = mask + (size_t)b * Tc * Tc;

    // Q fragments: 2 M-tiles x 2 dk-chunks ([0,32), [32,64))
    v16bf qf[2][2];
#pragma unroll
    for (int i = 0; i < 2; ++i) {
        const bf16_t* qrow = Qb + (size_t)(tq0 + 16 * i + ln) * DKc;
        qf[i][0] = load_afrag(qrow, hi);
        qf[i][1] = load_afrag(qrow + 32, hi);
    }

    // all-ones B fragment: P @ ones(32x16) == row sums broadcast to all cols
    v16bf ones;
#pragma unroll
    for (int e = 0; e < 16; ++e) ones[e] = (bf16_t)1.0f;

    float mrun[2][8];
#pragma unroll
    for (int i = 0; i < 2; ++i)
#pragma unroll
        for (int r = 0; r < 8; ++r) mrun[i][r] = -3.0e38f;
    v8f acc[2][4] = {{v8f{}, v8f{}, v8f{}, v8f{}}, {v8f{}, v8f{}, v8f{}, v8f{}}};
    v8f rsum[2]   = {v8f{}, v8f{}};

    const float sc = 0.125f;  // 1/sqrt(DK)
    bf16_t (*pbw)[16 * PROW] = pbuf[wslot];

    for (int kc = 0; kc < Tc; kc += 32) {
        if (kc + 32 < Tc)
            __builtin_prefetch(Kb + (size_t)(kc + 32 + ln) * DKc, 0, 0);

        // ---- K fragments for this chunk: 2 key-halves x 2 dk-chunks ----
        v16bf kf[2][2];
        {
            const bf16_t* kr0 = Kb + (size_t)(kc + ln) * DKc;
            const bf16_t* kr1 = Kb + (size_t)(kc + 16 + ln) * DKc;
            kf[0][0] = load_bfrag(kr0, hi);  kf[0][1] = load_bfrag(kr0 + 32, hi);
            kf[1][0] = load_bfrag(kr1, hi);  kf[1][1] = load_bfrag(kr1 + 32, hi);
        }

        // ---- scores: S[i][t] = Q_i * K_t^T ----
        v8f s[2][2];
#pragma unroll
        for (int i = 0; i < 2; ++i)
#pragma unroll
            for (int t = 0; t < 2; ++t) {
                v8f z = v8f{};
                z = wmma_bf16(qf[i][0], kf[t][0], z);
                z = wmma_bf16(qf[i][1], kf[t][1], z);
                s[i][t] = z;
            }

        // ---- scale + additive mask ----
#pragma unroll
        for (int i = 0; i < 2; ++i)
#pragma unroll
            for (int r = 0; r < 8; ++r) {
                const int tq = tq0 + 16 * i + r + 8 * hi;
                const float* mr = Mb + (size_t)tq * Tc + kc;
                s[i][0][r] = s[i][0][r] * sc + mr[ln];
                s[i][1][r] = s[i][1][r] * sc + mr[16 + ln];
            }

        // ---- online softmax: running max (shuffle reduce), rescale ----
#pragma unroll
        for (int i = 0; i < 2; ++i)
#pragma unroll
            for (int r = 0; r < 8; ++r) {
                float mx = fmaxf(s[i][0][r], s[i][1][r]);
#pragma unroll
                for (int o = 1; o < 16; o <<= 1)
                    mx = fmaxf(mx, __shfl_xor(mx, o, 32));
                const float mnew = fmaxf(mrun[i][r], mx);
                const float corr = __expf(mrun[i][r] - mnew);
                mrun[i][r] = mnew;
                s[i][0][r] = __expf(s[i][0][r] - mnew);
                s[i][1][r] = __expf(s[i][1][r] - mnew);
                rsum[i][r] *= corr;
#pragma unroll
                for (int j = 0; j < 4; ++j) acc[i][j][r] *= corr;
            }

        // ---- stage P tiles (16x32 bf16) through LDS: D-layout -> A-frag ----
#pragma unroll
        for (int i = 0; i < 2; ++i)
#pragma unroll
            for (int r = 0; r < 8; ++r) {
                const int rowm = r + 8 * hi;
                pbw[i][rowm * PROW + ln]      = (bf16_t)s[i][0][r];
                pbw[i][rowm * PROW + 16 + ln] = (bf16_t)s[i][1][r];
            }
        const v16bf pf0 = load_afrag(&pbw[0][ln * PROW], hi);
        const v16bf pf1 = load_afrag(&pbw[1][ln * PROW], hi);

        // ---- ctx += P * V (V fragments shared by both M-tiles) ----
#pragma unroll
        for (int j = 0; j < 4; ++j) {
            const int dk = j * 16 + ln;
            v16bf vf = load_bfrag(Vb + (size_t)dk * Tc + kc, hi);
            acc[0][j] = wmma_bf16(pf0, vf, acc[0][j]);
            acc[1][j] = wmma_bf16(pf1, vf, acc[1][j]);
        }
        // ---- row sums via matrix pipe ----
        rsum[0] = wmma_bf16(pf0, ones, rsum[0]);
        rsum[1] = wmma_bf16(pf1, ones, rsum[1]);
    }

    // ---- normalize and store ctx as [B,T,D] bf16 ----
#pragma unroll
    for (int i = 0; i < 2; ++i)
#pragma unroll
        for (int r = 0; r < 8; ++r) {
            const float inv = 1.0f / rsum[i][r];
            const int t = tq0 + 16 * i + r + 8 * hi;
            bf16_t* crow = CTX + ((size_t)b * Tc + t) * Dc + h * DKc;
#pragma unroll
            for (int j = 0; j < 4; ++j)
                crow[j * 16 + ln] = (bf16_t)(acc[i][j][r] * inv);
        }
}

// ---------------------------------------------------------------------------
// Host-side orchestration
// ---------------------------------------------------------------------------
extern "C" void kernel_launch(void* const* d_in, const int* in_sizes, int n_in,
                              void* d_out, int out_size, void* d_ws, size_t ws_size,
                              hipStream_t stream) {
    const float* q  = (const float*)d_in[0];
    const float* k  = (const float*)d_in[1];
    const float* v  = (const float*)d_in[2];
    const float* mk = (const float*)d_in[3];
    const float* Wq = (const float*)d_in[4];  const float* bq = (const float*)d_in[5];
    const float* Wk = (const float*)d_in[6];  const float* bk = (const float*)d_in[7];
    const float* Wv = (const float*)d_in[8];  const float* bv = (const float*)d_in[9];
    const float* Wo = (const float*)d_in[10]; const float* bo = (const float*)d_in[11];
    float* out = (float*)d_out;

    const size_t nX = (size_t)Bc * Tc * Dc;   // 8M elems
    const size_t nW = (size_t)Dc * Dc;        // 1M elems

    char* ws = (char*)d_ws;
    bf16_t* Xq  = (bf16_t*)ws;                 ws += nX * 2;
    bf16_t* Xk  = (bf16_t*)ws;                 ws += nX * 2;
    bf16_t* Xv  = (bf16_t*)ws;                 ws += nX * 2;
    bf16_t* Wqh = (bf16_t*)ws;                 ws += nW * 2;
    bf16_t* Wkh = (bf16_t*)ws;                 ws += nW * 2;
    bf16_t* Wvh = (bf16_t*)ws;                 ws += nW * 2;
    bf16_t* Woh = (bf16_t*)ws;                 ws += nW * 2;
    bf16_t* Qh  = (bf16_t*)ws;                 ws += nX * 2;  // [B,H,T,DK]
    bf16_t* Kh  = (bf16_t*)ws;                 ws += nX * 2;  // [B,H,T,DK]
    bf16_t* VTh = (bf16_t*)ws;                 ws += nX * 2;  // [B,H,DK,T]
    bf16_t* CTX = (bf16_t*)ws;                 ws += nX * 2;  // [B,T,D]

    // 1) fp32 -> bf16 conversions
    {
        const int n4x = (int)(nX / 4), n4w = (int)(nW / 4);
        cvt_bf16_kernel<<<n4x / 256, 256, 0, stream>>>(q,  Xq,  n4x);
        cvt_bf16_kernel<<<n4x / 256, 256, 0, stream>>>(k,  Xk,  n4x);
        cvt_bf16_kernel<<<n4x / 256, 256, 0, stream>>>(v,  Xv,  n4x);
        cvt_bf16_kernel<<<n4w / 256, 256, 0, stream>>>(Wq, Wqh, n4w);
        cvt_bf16_kernel<<<n4w / 256, 256, 0, stream>>>(Wk, Wkh, n4w);
        cvt_bf16_kernel<<<n4w / 256, 256, 0, stream>>>(Wv, Wvh, n4w);
        cvt_bf16_kernel<<<n4w / 256, 256, 0, stream>>>(Wo, Woh, n4w);
    }

    // 2) projections: 4096 wave-tiles (32x64 each), 8 waves / 256-thread block
    const int pblocks = (Bc * Tc / 32) * (Dc / 64) / 8;  // 512
    proj_gemm_kernel<0><<<pblocks, 256, 0, stream>>>(Xq, Wqh, bq, Qh);
    proj_gemm_kernel<0><<<pblocks, 256, 0, stream>>>(Xk, Wkh, bk, Kh);
    proj_gemm_kernel<1><<<pblocks, 256, 0, stream>>>(Xv, Wvh, bv, VTh);

    // 3) flash attention: B*H*(T/32) = 4096 waves, 4 waves / 128-thread block
    flash_attn_kernel<<<(Bc * Hc * (Tc / 32)) / 4, 128, 0, stream>>>(
        Qh, Kh, VTh, mk, CTX);

    // 4) output projection -> f32 d_out
    proj_gemm_kernel<2><<<pblocks, 256, 0, stream>>>(CTX, Woh, bo, out);
}
